// PoemRNN_10479720202856
// MI455X (gfx1250) — compile-verified, matrix operand order checked by reading, not run on hardware
//
#include <hip/hip_runtime.h>
#include <hip/hip_bf16.h>
#include <cstdint>

typedef __bf16 bf16;
typedef bf16  bf16x4  __attribute__((ext_vector_type(4)));
typedef bf16  bf16x8  __attribute__((ext_vector_type(8)));
typedef bf16  bf16x16 __attribute__((ext_vector_type(16)));
typedef float f32x4   __attribute__((ext_vector_type(4)));
typedef float f32x8   __attribute__((ext_vector_type(8)));

#define B_SZ 1024
#define T_SZ 256
#define E_SZ 100
#define H_SZ 128
#define V_SZ 32000

// 16x32 bf16 WMMA fragment from LDS tile (row-major, K-stride 128):
// lanes 0-15 K {0..7,16..23}, lanes 16-31 K {8..15,24..31} -> 2x ds_load_b128.
__device__ __forceinline__ bf16x16 frag_lds(const bf16* base, int row, int kb, int lane) {
  const bf16* p = base + row * H_SZ + kb * 32 + ((lane >> 4) & 1) * 8;
  bf16x8 lo = *(const bf16x8*)(p);
  bf16x8 hi = *(const bf16x8*)(p + 16);
  return __builtin_shufflevector(lo, hi, 0,1,2,3,4,5,6,7,8,9,10,11,12,13,14,15);
}

__device__ __forceinline__ f32x8 wmma_bf16(bf16x16 a, bf16x16 b, f32x8 c) {
  return __builtin_amdgcn_wmma_f32_16x16x32_bf16(false, a, false, b, (short)0, c, false, false);
}

// f32x4 -> bf16x4 convert + single 8B LDS store
__device__ __forceinline__ void cvt_store4(bf16* dst, f32x4 v) {
  bf16x4 o;
  o.x = (bf16)v.x; o.y = (bf16)v.y; o.z = (bf16)v.z; o.w = (bf16)v.w;
  *(bf16x4*)dst = o;
}

// Hardware tanh (TRANS op). v_nop after it covers the TRANS result-use hazard
// the compiler cannot track through inline asm (ISA 7.4 rule 3).
__device__ __forceinline__ float htanh(float x) {
  float y;
  asm volatile("v_tanh_f32 %0, %1\n\tv_nop" : "=v"(y) : "v"(x));
  return y;
}

// ---------------------------------------------------------------------------
// Kernel 1: fused embedding gather + input projection
//   xp[t][b][h] = emb[x[b,t]] . W_ih^T + (b_ih + b_hh)
// ---------------------------------------------------------------------------
__global__ __launch_bounds__(256) void proj_kernel(const int* __restrict__ x,
                                                   const float* __restrict__ emb,
                                                   const float* __restrict__ W_ih,
                                                   const float* __restrict__ b_ih,
                                                   const float* __restrict__ b_hh,
                                                   float* __restrict__ xp) {
  __shared__ bf16 sA[64 * 128];   // gathered embeddings, K padded 100->128
  __shared__ bf16 sW[128 * 128];  // W_ih [N=H][K=E pad]
  const int tid  = threadIdx.x;
  const int row0 = blockIdx.x * 64;  // base bt row

  // W_ih rows are 400B (16B aligned): float4 loads, zero-pad K 100->128
  for (int idx = tid; idx < 128 * 32; idx += 256) {
    int r = idx >> 5, k4 = idx & 31;
    f32x4 v = {};
    if (k4 < 25) v = *(const f32x4*)(W_ih + r * E_SZ + k4 * 4);
    cvt_store4(&sW[r * H_SZ + k4 * 4], v);
  }
  for (int idx = tid; idx < 64 * 32; idx += 256) {
    int r = idx >> 5, k4 = idx & 31;
    int tok = x[row0 + r];
    f32x4 v = {};
    if (k4 < 25) v = *(const f32x4*)(emb + (size_t)tok * E_SZ + k4 * 4);
    cvt_store4(&sA[r * H_SZ + k4 * 4], v);
  }
  __syncthreads();

  const int wave = tid >> 5, lane = tid & 31;
  const int ln = lane & 15, hb = (lane >> 4) & 1;
  const int mt  = wave & 3;
  const int ntg = (wave >> 2) * 4;

  bf16x16 a[4];
#pragma unroll
  for (int kb = 0; kb < 4; ++kb) a[kb] = frag_lds(sA, mt * 16 + ln, kb, lane);

#pragma unroll
  for (int ni = 0; ni < 4; ++ni) {
    const int col = (ntg + ni) * 16 + ln;
    f32x8 c = {};
#pragma unroll
    for (int kb = 0; kb < 4; ++kb) {
      bf16x16 b = frag_lds(sW, col, kb, lane);
      c = wmma_bf16(a[kb], b, c);
    }
    const float bias = b_ih[col] + b_hh[col];
#pragma unroll
    for (int i = 0; i < 8; ++i) {
      int rg = row0 + mt * 16 + i + hb * 8;   // bt index
      int t  = rg & (T_SZ - 1);               // T = 256 (pow2)
      int b  = rg >> 8;
      xp[((size_t)t * B_SZ + b) * H_SZ + col] = c[i] + bias;
    }
  }
}

// ---------------------------------------------------------------------------
// Kernel 2: recurrence. WG g owns batch rows [16g,16g+16) for all T steps.
// W_hh B-fragments hoisted to registers (loop-invariant); h lives in LDS as
// bf16; xp[t] double-buffered in registers as the WMMA C-init.
// ---------------------------------------------------------------------------
__global__ __launch_bounds__(256) void scan_kernel(const float* __restrict__ W_hh,
                                                   const float* __restrict__ xp,
                                                   bf16* __restrict__ h_out) {
  __shared__ bf16 sW[128 * 128];  // W_hh [N=i][K=j]
  __shared__ bf16 sH[16 * 128];   // h tile [16 rows][K=128]
  const int tid = threadIdx.x;

  for (int idx = tid; idx < (128 * 128) / 4; idx += 256)
    cvt_store4(&sW[idx * 4], *(const f32x4*)(W_hh + idx * 4));
  for (int idx = tid; idx < 16 * 128; idx += 256) sH[idx] = (bf16)0.0f;

  const int wave = tid >> 5, lane = tid & 31;
  const int ln = lane & 15, hb = (lane >> 4) & 1;
  const int col   = wave * 16 + ln;
  const int brow0 = blockIdx.x * 16;

  float cur[8], nxt[8];
#pragma unroll
  for (int i = 0; i < 8; ++i)
    cur[i] = xp[(size_t)(brow0 + i + hb * 8) * H_SZ + col];  // t = 0
  __syncthreads();

  // loop-invariant W_hh fragments (32 VGPRs) -> per-step LDS reads halved
  bf16x16 bw[4];
#pragma unroll
  for (int kb = 0; kb < 4; ++kb) bw[kb] = frag_lds(sW, col, kb, lane);

  for (int t = 0; t < T_SZ; ++t) {
    if (t + 1 < T_SZ) {  // prefetch next timestep's accumulator init
#pragma unroll
      for (int i = 0; i < 8; ++i)
        nxt[i] = xp[((size_t)(t + 1) * B_SZ + brow0 + i + hb * 8) * H_SZ + col];
    }
    f32x8 c;
#pragma unroll
    for (int i = 0; i < 8; ++i) c[i] = cur[i];
#pragma unroll
    for (int kb = 0; kb < 4; ++kb) {
      bf16x16 a = frag_lds(sH, ln, kb, lane);   // all waves read full h
      c = wmma_bf16(a, bw[kb], c);
    }
    __syncthreads();  // all reads of sH done before overwrite
#pragma unroll
    for (int i = 0; i < 8; ++i)
      sH[(i + hb * 8) * H_SZ + col] = (bf16)htanh(c[i]);
    __syncthreads();  // writes visible before next step's reads
#pragma unroll
    for (int i = 0; i < 8; ++i) cur[i] = nxt[i];
  }

  for (int idx = tid; idx < 16 * 128; idx += 256)
    h_out[(size_t)brow0 * H_SZ + idx] = sH[idx];
}

// ---------------------------------------------------------------------------
// Kernel 3: head. out[b][v] = h_last . W_fc^T + b_fc.
// h tile is already bf16 -> staged via CDNA5 async DMA to LDS (ASYNCcnt path).
// ---------------------------------------------------------------------------
__global__ __launch_bounds__(256) void head_kernel(const bf16* __restrict__ h,
                                                   const float* __restrict__ W_fc,
                                                   const float* __restrict__ b_fc,
                                                   float* __restrict__ out) {
  __shared__ bf16 sH[64 * 128];   // 16 KB
  __shared__ bf16 sW[128 * 128];  // 32 KB
  const int tid = threadIdx.x;
  const int n0  = blockIdx.x * 128;
  const int m0  = blockIdx.y * 64;

  // Async-DMA the 16KB bf16 h tile straight into LDS (GVS: SGPR base + VGPR
  // offset). Generic->32-bit truncation of a __shared__ pointer yields the
  // LDS byte address (ISA 10.2 LDS aperture mapping).
  {
    const bf16* gsrc = h + (size_t)m0 * H_SZ;
    const unsigned lbase = (unsigned)(uintptr_t)(&sH[0]);
#pragma unroll
    for (int i = 0; i < 4; ++i) {
      unsigned goff = tid * 16 + i * 4096;
      unsigned loff = lbase + goff;
      asm volatile("global_load_async_to_lds_b128 %0, %1, %2"
                   :: "v"(loff), "v"(goff), "s"(gsrc) : "memory");
    }
  }
  // W_fc tile needs f32->bf16 conversion: regular vector loads overlap the DMA
  for (int idx = tid; idx < (128 * 128) / 4; idx += 256)
    cvt_store4(&sW[idx * 4], *(const f32x4*)(W_fc + (size_t)n0 * H_SZ + idx * 4));

  asm volatile("s_wait_asynccnt 0x0" ::: "memory");
  __syncthreads();

  const int wave = tid >> 5, lane = tid & 31;
  const int ln = lane & 15, hb = (lane >> 4) & 1;
  const int nloc = wave * 16 + ln;
  const int col  = n0 + nloc;
  const float bias = b_fc[col];

  bf16x16 bfr[4];
#pragma unroll
  for (int kb = 0; kb < 4; ++kb) bfr[kb] = frag_lds(sW, nloc, kb, lane);

#pragma unroll
  for (int mt = 0; mt < 4; ++mt) {
    f32x8 c = {};
#pragma unroll
    for (int kb = 0; kb < 4; ++kb) {
      bf16x16 a = frag_lds(sH, mt * 16 + ln, kb, lane);
      c = wmma_bf16(a, bfr[kb], c);
    }
#pragma unroll
    for (int i = 0; i < 8; ++i) {
      int m = m0 + mt * 16 + i + hb * 8;
      out[(size_t)m * V_SZ + col] = c[i] + bias;
    }
  }
}

extern "C" void kernel_launch(void* const* d_in, const int* in_sizes, int n_in,
                              void* d_out, int out_size, void* d_ws, size_t ws_size,
                              hipStream_t stream) {
  const int*   x    = (const int*)d_in[0];
  const float* emb  = (const float*)d_in[1];
  const float* W_ih = (const float*)d_in[2];
  const float* W_hh = (const float*)d_in[3];
  const float* b_ih = (const float*)d_in[4];
  const float* b_hh = (const float*)d_in[5];
  const float* W_fc = (const float*)d_in[6];
  const float* b_fc = (const float*)d_in[7];
  float* out = (float*)d_out;

  float* xp     = (float*)d_ws;                                  // [T][B][H] f32: 128 MB
  bf16*  h_last = (bf16*)((char*)d_ws + (size_t)T_SZ * B_SZ * H_SZ * sizeof(float));

  proj_kernel<<<(B_SZ * T_SZ) / 64, 256, 0, stream>>>(x, emb, W_ih, b_ih, b_hh, xp);
  scan_kernel<<<B_SZ / 16, 256, 0, stream>>>(W_hh, xp, h_last);
  head_kernel<<<dim3(V_SZ / 128, B_SZ / 64), 256, 0, stream>>>(h_last, W_fc, b_fc, out);
}
